// GRUTimeSeriesBinaryClassifier_74062416052895
// MI455X (gfx1250) — matrix operand order repaired and checked by reading, not compiled, not run on hardware
//
#include <hip/hip_runtime.h>

typedef __attribute__((ext_vector_type(16))) _Float16 v16h;
typedef __attribute__((ext_vector_type(8)))  _Float16 v8h;
typedef __attribute__((ext_vector_type(8)))  float    v8f;

#define GRU_B   256
#define GRU_T   2048
#define GRU_DIN 32
#define GRU_H   64
#define HSTR    72   // padded LDS row stride (halves): 144B rows, conflict-free across 16 rows

// ---- fast transcendentals: v_exp_f32 (2^x) + v_rcp_f32, no IEEE-div expansion ----
static __device__ __forceinline__ float fast_exp(float x) {
  return __builtin_amdgcn_exp2f(x * 1.4426950408889634f);
}
static __device__ __forceinline__ float fast_rcp(float x) { return __builtin_amdgcn_rcpf(x); }
static __device__ __forceinline__ float sigm(float x)   { return fast_rcp(1.f + fast_exp(-x)); }
static __device__ __forceinline__ float tanh_f(float x) { return 1.f - 2.f * fast_rcp(1.f + fast_exp(2.f * x)); }

static __device__ __forceinline__ v8f wmma16(v16h a, v16h b, v8f c) {
  return __builtin_amdgcn_wmma_f32_16x16x32_f16(false, a, false, b, (short)0, c, false, false);
}

static __device__ __forceinline__ v8f splat8(float v) {
  v8f r;
#pragma unroll
  for (int i = 0; i < 8; ++i) r[i] = v;
  return r;
}

// A-matrix f16 16x32 fragment from LDS h buffer (two 16B runs -> 2x ds_load_b128).
// row = lane&15; kbase = 32*frag + 8*(lane>=16); halves 0..7 -> K kbase.., 8..15 -> K kbase+16..
static __device__ __forceinline__ v16h load_a_lds(const _Float16* hbuf, int row, int kbase) {
  const _Float16* p = hbuf + row * HSTR + kbase;
  const v8h lo = *(const v8h*)(p);
  const v8h hi = *(const v8h*)(p + 16);
  v16h a;
#pragma unroll
  for (int i = 0; i < 8; ++i) { a[i] = lo[i]; a[i + 8] = hi[i]; }
  return a;
}

// A fragment from global f32 x (K = DIN = 32), cvt to f16. koff = 8*(lane>=16).
static __device__ __forceinline__ v16h load_a_x(const float* xrow, int koff) {
  v16h a;
#pragma unroll
  for (int i = 0; i < 8; ++i) {
    a[i]     = (_Float16)xrow[koff + i];
    a[i + 8] = (_Float16)xrow[koff + 16 + i];
  }
  return a;
}

// B-matrix f16 32x16 fragment from f32 weight W[n][K] (GEMM is in @ W^T -> B[k][n] = W[n][k]).
// col = lane&15; halves 0..7 -> K = kbase + i, 8..15 -> K = kbase + 16 + i; kbase = 32*kchunk + 8*(lane>=16).
static __device__ __forceinline__ v16h load_b_w(const float* W, int ldw, int nrow, int kbase) {
  const float* p = W + (size_t)nrow * ldw + kbase;
  v16h b;
#pragma unroll
  for (int i = 0; i < 8; ++i) { b[i] = (_Float16)p[i]; b[i + 8] = (_Float16)p[i + 16]; }
  return b;
}

// Store C/D-layout f32 tile as f16 into LDS h buffer. VGPR r -> M = r + 8*(lane>=16), N = lane&15.
static __device__ __forceinline__ void store_h_lds(_Float16* hbuf, const v8f& h, int colBase, int lane) {
  const int c = colBase + (lane & 15);
  const int rbase = (lane >> 4) * 8;
#pragma unroll
  for (int r = 0; r < 8; ++r) hbuf[(rbase + r) * HSTR + c] = (_Float16)h[r];
}

// 8 waves: waves 0-3 = layer0 pipeline stage, waves 4-7 = layer1 stage (one step behind).
__global__ __launch_bounds__(256)
void gru2_fused_kernel(const float* __restrict__ x,
                       const float* __restrict__ Wih0, const float* __restrict__ Whh0,
                       const float* __restrict__ bih0, const float* __restrict__ bhh0,
                       const float* __restrict__ Wih1, const float* __restrict__ Whh1,
                       const float* __restrict__ bih1, const float* __restrict__ bhh1,
                       const float* __restrict__ fc1w, const float* __restrict__ fc1b,
                       const float* __restrict__ outw, const float* __restrict__ outb,
                       float* __restrict__ out) {
  __shared__ _Float16 h0s[2][16 * HSTR];  // layer0 hidden state, f16, ping-pong
  __shared__ _Float16 h1s[2][16 * HSTR];  // layer1 hidden state, f16, ping-pong
  __shared__ float    hfin[16 * GRU_H];   // final h_last, f32

  const int lane   = threadIdx.x & 31;
  // Broadcast the wave id into an SGPR so the stage split compiles to scalar
  // s_cmp/s_cbranch (EXEC untouched -> no saveexec per step, full EXEC for WMMA).
  const int waveId = __builtin_amdgcn_readfirstlane((int)(threadIdx.x >> 5));  // 0..7
  const bool isA   = (waveId < 4);        // A = layer0 stage, B = layer1 stage
  const int cg     = waveId & 3;          // owned hidden-column group within the stage
  const int laneLo = lane & 15;
  const int laneHi = lane >> 4;
  const int koff   = laneHi * 8;          // A-frag K offset (halves)
  const int kbw    = laneHi * 8;          // B-frag K offset
  const int col    = cg * 16 + laneLo;    // owned hidden column (0..63)

  // Per-stage biases (r,z fold b_ih+b_hh; n keeps them split: n = tanh(xn + r*hn)).
  float bR, bZ, bNx, bNh;
  if (isA) {
    bR  = bih0[col]       + bhh0[col];
    bZ  = bih0[64 + col]  + bhh0[64 + col];
    bNx = bih0[128 + col];
    bNh = bhh0[128 + col];
  } else {
    bR  = bih1[col]       + bhh1[col];
    bZ  = bih1[64 + col]  + bhh1[64 + col];
    bNx = bih1[128 + col];
    bNh = bhh1[128 + col];
  }
  // Loop-invariant accumulator seeds: first WMMA of each chain consumes these as C,
  // so no per-step splat/v_mov init is needed.
  const v8f vbR  = splat8(bR);
  const v8f vbZ  = splat8(bZ);
  const v8f vbNx = splat8(bNx);
  const v8f vbNh = splat8(bNh);

  // Resident weight fragments: only this stage's layer -> ~72-96 VGPRs (stays under 256 total).
  v16h wx0[3];              // layer0 input weights, K=32 (A waves only)
  v16h wxh[3][2];           // A: Whh0 ; B: Wih1   (K=64)
  v16h whh[3][2];           // A: unused dup of wxh; B: Whh1
#pragma unroll
  for (int g = 0; g < 3; ++g) {
    const int n = g * 64 + col;
    if (isA) {
      wx0[g] = load_b_w(Wih0, GRU_DIN, n, kbw);
#pragma unroll
      for (int kc = 0; kc < 2; ++kc) {
        wxh[g][kc] = load_b_w(Whh0, GRU_H, n, kc * 32 + kbw);
        whh[g][kc] = wxh[g][kc];  // unused on A path
      }
    } else {
#pragma unroll
      for (int kc = 0; kc < 2; ++kc) {
        wxh[g][kc] = load_b_w(Wih1, GRU_H, n, kc * 32 + kbw);
        whh[g][kc] = load_b_w(Whh1, GRU_H, n, kc * 32 + kbw);
      }
    }
  }

  // Zero all state buffers (h(-1) = 0).
  for (int i = threadIdx.x; i < 16 * HSTR; i += 256) {
    h0s[0][i] = (_Float16)0.f; h0s[1][i] = (_Float16)0.f;
    h1s[0][i] = (_Float16)0.f; h1s[1][i] = (_Float16)0.f;
  }
  v8f hp = splat8(0.f);   // this stage's f32 copy of its owned h tile
  __syncthreads();

  const float* xrow = x + (size_t)(blockIdx.x * 16 + laneLo) * GRU_T * GRU_DIN;
  v16h axc = load_a_x(xrow, koff);  // prefetched x(0) A-frag (A waves use it)

  // Pipeline: iter k -> A computes h0(k) [k<T], B computes h1(k-1) [k>=1]. One barrier per step.
  for (int k = 0; k <= GRU_T; ++k) {
    const int rb = (k + 1) & 1;   // read buffer  (state from iter k-1)
    const int wb = k & 1;         // write buffer

    if (isA) {
      if (k < GRU_T) {
        // prefetch next x A-frag (clamped; latency hidden behind WMMAs + barrier)
        const int kn = (k + 1 < GRU_T) ? (k + 1) : (GRU_T - 1);
        const v16h axn = load_a_x(xrow + (size_t)kn * GRU_DIN, koff);

        v8f aR  = wmma16(axc, wx0[0], vbR);
        v8f aZ  = wmma16(axc, wx0[1], vbZ);
        v8f aNx = wmma16(axc, wx0[2], vbNx);

        const v16h ah0 = load_a_lds(h0s[rb], laneLo, koff);        // h0(k-1)
        const v16h ah1 = load_a_lds(h0s[rb], laneLo, 32 + koff);
        v8f aNh = wmma16(ah0, wxh[2][0], vbNh);
        aR  = wmma16(ah0, wxh[0][0], aR);  aR  = wmma16(ah1, wxh[0][1], aR);
        aZ  = wmma16(ah0, wxh[1][0], aZ);  aZ  = wmma16(ah1, wxh[1][1], aZ);
        aNh = wmma16(ah1, wxh[2][1], aNh);

        v8f hn;
#pragma unroll
        for (int r = 0; r < 8; ++r) {
          const float rr = sigm(aR[r]);
          const float zz = sigm(aZ[r]);
          const float nn = tanh_f(aNx[r] + rr * aNh[r]);
          hn[r] = nn + zz * (hp[r] - nn);       // (1-z)*n + z*h
        }
        hp = hn;
        store_h_lds(h0s[wb], hn, cg * 16, lane);
        axc = axn;
      }
    } else {
      if (k >= 1) {
        const v16h a0 = load_a_lds(h0s[rb], laneLo, koff);         // layer1 input = h0(k-1)
        const v16h a1 = load_a_lds(h0s[rb], laneLo, 32 + koff);
        const v16h b0 = load_a_lds(h1s[rb], laneLo, koff);         // h1(k-2)
        const v16h b1 = load_a_lds(h1s[rb], laneLo, 32 + koff);

        v8f cR  = wmma16(a0, wxh[0][0], vbR);
        v8f cZ  = wmma16(a0, wxh[1][0], vbZ);
        v8f cNx = wmma16(a0, wxh[2][0], vbNx);
        v8f cNh = wmma16(b0, whh[2][0], vbNh);
        cR  = wmma16(a1, wxh[0][1], cR);
        cZ  = wmma16(a1, wxh[1][1], cZ);
        cNx = wmma16(a1, wxh[2][1], cNx);
        cNh = wmma16(b1, whh[2][1], cNh);
        cR  = wmma16(b0, whh[0][0], cR);  cR  = wmma16(b1, whh[0][1], cR);
        cZ  = wmma16(b0, whh[1][0], cZ);  cZ  = wmma16(b1, whh[1][1], cZ);

        v8f hn;
#pragma unroll
        for (int r = 0; r < 8; ++r) {
          const float rr = sigm(cR[r]);
          const float zz = sigm(cZ[r]);
          const float nn = tanh_f(cNx[r] + rr * cNh[r]);
          hn[r] = nn + zz * (hp[r] - nn);
        }
        hp = hn;
        store_h_lds(h1s[wb], hn, cg * 16, lane);
      }
    }
    __syncthreads();   // single barrier per timestep (ping-pong buffers)
  }

  // B waves publish final f32 h_last = h1(T-1).
  if (!isA) {
    const int rbase = laneHi * 8;
#pragma unroll
    for (int r = 0; r < 8; ++r) hfin[(rbase + r) * GRU_H + col] = hp[r];
  }
  __syncthreads();

  // Head: out = relu(h @ fc1^T + b1) @ outw^T + outb -- tiny, scalar per batch row.
  if (threadIdx.x < 16) {
    const int row = threadIdx.x;
    float o = outb[0];
    for (int j = 0; j < 32; ++j) {
      float a = fc1b[j];
#pragma unroll 8
      for (int kk = 0; kk < GRU_H; ++kk) a += hfin[row * GRU_H + kk] * fc1w[j * GRU_H + kk];
      o += fmaxf(a, 0.f) * outw[j];
    }
    out[blockIdx.x * 16 + row] = o;
  }
}

extern "C" void kernel_launch(void* const* d_in, const int* in_sizes, int n_in,
                              void* d_out, int out_size, void* d_ws, size_t ws_size,
                              hipStream_t stream) {
  (void)in_sizes; (void)n_in; (void)d_ws; (void)ws_size; (void)out_size;
  const float* x    = (const float*)d_in[0];
  const float* Wih0 = (const float*)d_in[1];
  const float* Whh0 = (const float*)d_in[2];
  const float* bih0 = (const float*)d_in[3];
  const float* bhh0 = (const float*)d_in[4];
  const float* Wih1 = (const float*)d_in[5];
  const float* Whh1 = (const float*)d_in[6];
  const float* bih1 = (const float*)d_in[7];
  const float* bhh1 = (const float*)d_in[8];
  const float* fc1w = (const float*)d_in[9];
  const float* fc1b = (const float*)d_in[10];
  const float* outw = (const float*)d_in[11];
  const float* outb = (const float*)d_in[12];

  gru2_fused_kernel<<<GRU_B / 16, 256, 0, stream>>>(
      x, Wih0, Whh0, bih0, bhh0, Wih1, Whh1, bih1, bhh1,
      fc1w, fc1b, outw, outb, (float*)d_out);
}
